// RetNet_50809463111763
// MI455X (gfx1250) — matrix operand order repaired and checked by reading, not compiled, not run on hardware
//
#include <hip/hip_runtime.h>
#include <hip/hip_bf16.h>

// ---------------------------------------------------------------------------
// RetNet block for MI455X (gfx1250): bf16 WMMA + async global->LDS staging.
// ---------------------------------------------------------------------------

#define EMBED 1024
#define VDIM  2048
#define NH    8
#define KD    128     // key dim per head
#define HD    256     // value dim per head
#define BATCH 2
#define SEQ   2048
#define MROWS (BATCH*SEQ)   // 4096

typedef __attribute__((ext_vector_type(16))) __bf16 bf16x16;
typedef __attribute__((ext_vector_type(8)))  float  f32x8;
typedef __attribute__((ext_vector_type(4)))  int    i32x4;   // trivial 16B chunk

union BF16x16 { i32x4 q[2]; bf16x16 v; };

#if defined(__AMDGCN__) && __has_builtin(__builtin_amdgcn_global_load_async_to_lds_b128) && __has_builtin(__builtin_amdgcn_s_wait_asynccnt)
#define USE_ASYNC_LDS 1
#else
#define USE_ASYNC_LDS 0
#endif

#if USE_ASYNC_LDS
typedef __attribute__((address_space(1))) i32x4 as1_i32x4;
typedef __attribute__((address_space(3))) i32x4 as3_i32x4;
__device__ inline as1_i32x4* gbl_cast(const void* p) {
    return (as1_i32x4*)(unsigned long long)p;
}
__device__ inline as3_i32x4* lds_cast(const void* p) {
    // flat LDS address: low 32 bits are the LDS offset on gfx1250
    return (as3_i32x4*)(unsigned int)(unsigned long long)p;
}
__device__ inline void async_cp16(const void* gsrc, void* ldst) {
    __builtin_amdgcn_global_load_async_to_lds_b128(gbl_cast(gsrc), lds_cast(ldst), 0, 0);
}
__device__ inline void async_wait0() { __builtin_amdgcn_s_wait_asynccnt(0); }
#else
__device__ inline void async_cp16(const void* gsrc, void* ldst) {
    *reinterpret_cast<i32x4*>(ldst) = *reinterpret_cast<const i32x4*>(gsrc);
}
__device__ inline void async_wait0() {}
#endif

__device__ inline unsigned short f2bf(float f) {
    unsigned int u = __float_as_uint(f);
    u += 0x7fffu + ((u >> 16) & 1u);          // round-to-nearest-even
    return (unsigned short)(u >> 16);
}
__device__ inline float bf2f(unsigned short h) {
    return __uint_as_float(((unsigned int)h) << 16);
}
__device__ inline f32x8 zero8() {
    f32x8 z;
#pragma unroll
    for (int i = 0; i < 8; ++i) z[i] = 0.0f;
    return z;
}
__device__ inline f32x8 wmma_bf16(const BF16x16& a, const BF16x16& b, f32x8 c) {
    return __builtin_amdgcn_wmma_f32_16x16x32_bf16(false, a.v, false, b.v,
                                                   (short)0, c, false, false);
}

// ---------------------------------------------------------------------------
// fp32 -> bf16 conversion kernels
// ---------------------------------------------------------------------------
__global__ void conv_bf16(const float* __restrict__ src,
                          unsigned short* __restrict__ dst, int n) {
    int i = blockIdx.x * blockDim.x + threadIdx.x;
    if (i < n) dst[i] = f2bf(src[i]);
}

// src is (K x N) row-major fp32 -> dst is (N x K) row-major bf16 (weight^T)
__global__ void conv_bf16_T(const float* __restrict__ src,
                            unsigned short* __restrict__ dst, int K, int N) {
    int i = blockIdx.x * blockDim.x + threadIdx.x;
    if (i < K * N) {
        int k = i / N, n = i - k * N;
        dst[(size_t)n * K + k] = f2bf(src[i]);
    }
}

// ---------------------------------------------------------------------------
// WMMA GEMM:  C(MxN) = A(MxK, bf16 rowmajor) * BT(NxK, bf16 rowmajor)^T
// 256 threads / 8 waves; block tile 128x128; k-step 32; double-buffered LDS.
// Wave (wm,wn) computes a 32x64 sub-tile: 2x4 WMMA tiles, operands preloaded.
// MODE 0: fp32 rowmajor store.  MODE 1: bf16 rowmajor store.
// MODE 2: bf16 scatter to V layout (b,h,d,s).
// ---------------------------------------------------------------------------
template <int MODE>
__global__ void __launch_bounds__(256)
gemm_bf16(const unsigned short* __restrict__ A,
          const unsigned short* __restrict__ BT,
          void* __restrict__ Cout, int M, int N, int K) {
    __shared__ __align__(16) unsigned short As[2 * 128 * 32];   // 16 KB
    __shared__ __align__(16) unsigned short Bs[2 * 128 * 32];   // 16 KB

    const int tid  = threadIdx.x;
    const int lane = tid & 31;
    const int w    = tid >> 5;
    const int wm   = w & 3;          // row group (32 rows each)
    const int wn   = w >> 2;         // col group (64 cols each)
    const int half = lane >> 4;
    const int l16  = lane & 15;
    const int mb   = blockIdx.y * 128;
    const int nb   = blockIdx.x * 128;

    f32x8 acc[2][4];
#pragma unroll
    for (int mt = 0; mt < 2; ++mt)
#pragma unroll
        for (int nt = 0; nt < 4; ++nt) acc[mt][nt] = zero8();

    // stage: copy A tile (128x32) + BT tile (128x32) into buffer `bf`
    auto stage = [&](int bf, int kb) {
#pragma unroll
        for (int it = 0; it < 2; ++it) {
            int idx = tid + it * 256;               // 512 chunks of 16B each
            int r = idx >> 2, c = (idx & 3) * 8;
            async_cp16(&A[(size_t)(mb + r) * K + kb + c],
                       &As[bf * 4096 + r * 32 + c]);
            async_cp16(&BT[(size_t)(nb + r) * K + kb + c],
                       &Bs[bf * 4096 + r * 32 + c]);
        }
    };

    stage(0, 0);
    async_wait0();
    __syncthreads();

    int buf = 0;
    for (int kb = 0; kb < K; kb += 32) {
        if (kb + 32 < K) stage(buf ^ 1, kb + 32);   // overlap with compute

        const unsigned short* Ab = &As[buf * 4096];
        const unsigned short* Bb = &Bs[buf * 4096];

        // Preload all operand fragments, then fire 8 WMMAs back-to-back.
        BF16x16 a[2], b[4];
#pragma unroll
        for (int mt = 0; mt < 2; ++mt) {
            int r = wm * 32 + mt * 16 + l16;
            a[mt].q[0] = *reinterpret_cast<const i32x4*>(&Ab[r * 32 + half * 8]);
            a[mt].q[1] = *reinterpret_cast<const i32x4*>(&Ab[r * 32 + 16 + half * 8]);
        }
#pragma unroll
        for (int nt = 0; nt < 4; ++nt) {
            int r = wn * 64 + nt * 16 + l16;
            b[nt].q[0] = *reinterpret_cast<const i32x4*>(&Bb[r * 32 + half * 8]);
            b[nt].q[1] = *reinterpret_cast<const i32x4*>(&Bb[r * 32 + 16 + half * 8]);
        }
#pragma unroll
        for (int mt = 0; mt < 2; ++mt)
#pragma unroll
            for (int nt = 0; nt < 4; ++nt)
                acc[mt][nt] = wmma_bf16(a[mt], b[nt], acc[mt][nt]);

        async_wait0();
        __syncthreads();
        buf ^= 1;
    }

#pragma unroll
    for (int mt = 0; mt < 2; ++mt) {
#pragma unroll
        for (int nt = 0; nt < 4; ++nt) {
#pragma unroll
            for (int v = 0; v < 8; ++v) {
                int row = mb + wm * 32 + mt * 16 + v + half * 8;
                int col = nb + wn * 64 + nt * 16 + l16;
                float val = acc[mt][nt][v];
                if (MODE == 0) {
                    ((float*)Cout)[(size_t)row * N + col] = val;
                } else if (MODE == 1) {
                    ((unsigned short*)Cout)[(size_t)row * N + col] = f2bf(val);
                } else {
                    // row = b*SEQ + s ; col = h*HD + d -> ((b*NH+h)*HD+d)*SEQ + s
                    int b = row >> 11, s = row & (SEQ - 1);
                    int h = col >> 8,  d = col & (HD - 1);
                    ((unsigned short*)Cout)[((size_t)((b * NH + h) * HD + d)) * SEQ + s] =
                        f2bf(val);
                }
            }
        }
    }
}

// ---------------------------------------------------------------------------
// Rotary embedding: q,k fp32 (b,s,EMBED) -> qr,kr bf16 (b,h,s,KD).
// Folds k * KD^-0.5.  One thread per (b,s,h,pair).
// ---------------------------------------------------------------------------
__global__ void rotary_kernel(const float* __restrict__ q,
                              const float* __restrict__ k,
                              unsigned short* __restrict__ qr,
                              unsigned short* __restrict__ kr) {
    int idx = blockIdx.x * blockDim.x + threadIdx.x;     // B*S*H*64 = 2^21
    int t = idx & 63;
    int h = (idx >> 6) & (NH - 1);
    int s = (idx >> 9) & (SEQ - 1);
    int b = idx >> 20;

    float ang = __powf(10000.0f, -(float)t / 63.0f);
    float th  = (float)s * ang;
    float sn, cs;
    __sincosf(th, &sn, &cs);

    size_t src = ((size_t)(b * SEQ + s)) * EMBED + h * KD + 2 * t;
    size_t dst = ((size_t)((b * NH + h) * SEQ + s)) * KD + 2 * t;

    float q0 = q[src], q1 = q[src + 1];
    qr[dst]     = f2bf(q0 * cs - q1 * sn);
    qr[dst + 1] = f2bf(q1 * cs + q0 * sn);

    const float kscale = 0.088388347648318447f;          // 128^-0.5
    float k0 = k[src] * kscale, k1 = k[src + 1] * kscale;
    kr[dst]     = f2bf(k0 * cs - k1 * sn);
    kr[dst + 1] = f2bf(k1 * cs + k0 * sn);
}

// ---------------------------------------------------------------------------
// Retention attention (flash-style, causal decay mask), fused with the
// denom normalization, per-head RMS norm and silu(g) gating.
// Block = 128 threads (4 waves) handles 16 query rows of one (b,h).
// K/V tiles double-buffered; next tile staged async during the PV WMMAs.
// Writes gated bf16 activations (b,s,VDIM), ready for the wo GEMM.
// ---------------------------------------------------------------------------
__global__ void __launch_bounds__(128)
retnet_attn(const unsigned short* __restrict__ qr,
            const unsigned short* __restrict__ kr,
            const unsigned short* __restrict__ vr,   // (b,h,d,s)
            const unsigned short* __restrict__ gbf,  // (b,s,VDIM)
            unsigned short* __restrict__ gated) {    // (b,s,VDIM)
    __shared__ __align__(16) unsigned short Qs[16 * 128];        //  4 KB
    __shared__ __align__(16) unsigned short Ks[2 * 64 * 128];    // 32 KB
    __shared__ __align__(16) unsigned short Vts[2 * 256 * 64];   // 64 KB (d-major)
    __shared__ __align__(16) unsigned short Ss[16 * 64];         //  2 KB
    __shared__ float rowsum[16];
    __shared__ float sumsq[16];

    const int tid  = threadIdx.x;
    const int lane = tid & 31;
    const int w    = tid >> 5;
    const int half = lane >> 4;
    const int l16  = lane & 15;
    const int bh   = blockIdx.y;            // b*NH + h
    const int h    = bh & (NH - 1);
    const int b    = bh >> 3;
    const int i0   = blockIdx.x * 16;
    const float decay = log1pf(-exp2f(-5.0f - (float)h));   // < 0

    // Q tile is fully contiguous in (b,h,s,KD) layout: 256 int4.
    const unsigned short* qbase = qr + ((size_t)bh * SEQ + i0) * KD;
#pragma unroll
    for (int it = 0; it < 2; ++it)
        reinterpret_cast<i32x4*>(Qs)[tid + it * 128] =
            reinterpret_cast<const i32x4*>(qbase)[tid + it * 128];
    if (tid < 16) { rowsum[tid] = 0.0f; sumsq[tid] = 0.0f; }

    f32x8 acc[4];
#pragma unroll
    for (int t = 0; t < 4; ++t) acc[t] = zero8();

    // stage K rows (contiguous) and V^T tile for j-tile jt into buffer bf
    auto stage = [&](int bf, int jt) {
        const int j0 = jt * 64;
        const unsigned short* kb_ = kr + ((size_t)bh * SEQ + j0) * KD;
#pragma unroll
        for (int it = 0; it < 8; ++it) {
            int idx = tid + it * 128;                // 1024 chunks
            async_cp16(&kb_[idx * 8], &Ks[bf * 8192 + idx * 8]);
        }
        const unsigned short* vb_ = vr + (size_t)bh * HD * SEQ + j0;
#pragma unroll
        for (int it = 0; it < 16; ++it) {
            int idx = tid + it * 128;                // 2048 chunks
            int d = idx >> 3, c = (idx & 7) * 8;
            async_cp16(&vb_[(size_t)d * SEQ + c], &Vts[bf * 16384 + d * 64 + c]);
        }
    };

    const int njt = (i0 >> 6) + 1;          // causal: j-tiles up to the diagonal
    stage(0, 0);
    async_wait0();
    __syncthreads();

    int buf = 0;
    for (int jt = 0; jt < njt; ++jt) {
        const int j0 = jt * 64;
        const unsigned short* Kb = &Ks[buf * 8192];
        const unsigned short* Vb = &Vts[buf * 16384];

        // Scores: wave w computes the 16x16 sub-tile at columns j0 + w*16.
        f32x8 D = zero8();
        {
            BF16x16 a[4], bm[4];
#pragma unroll
            for (int kc = 0; kc < 4; ++kc) {
                a[kc].q[0]  = *reinterpret_cast<const i32x4*>(&Qs[l16 * 128 + kc * 32 + half * 8]);
                a[kc].q[1]  = *reinterpret_cast<const i32x4*>(&Qs[l16 * 128 + kc * 32 + 16 + half * 8]);
                bm[kc].q[0] = *reinterpret_cast<const i32x4*>(&Kb[(w * 16 + l16) * 128 + kc * 32 + half * 8]);
                bm[kc].q[1] = *reinterpret_cast<const i32x4*>(&Kb[(w * 16 + l16) * 128 + kc * 32 + 16 + half * 8]);
            }
#pragma unroll
            for (int kc = 0; kc < 4; ++kc) D = wmma_bf16(a[kc], bm[kc], D);
        }
        // Apply causal decay mask, accumulate raw row sums, stash bf16 scores.
        const int jcol = j0 + w * 16 + l16;
#pragma unroll
        for (int v = 0; v < 8; ++v) {
            int i = i0 + v + half * 8;
            float sv = (jcol <= i) ? D[v] * __expf(decay * (float)(i - jcol)) : 0.0f;
            atomicAdd(&rowsum[v + half * 8], sv);
            Ss[(v + half * 8) * 64 + w * 16 + l16] = f2bf(sv);
        }
        __syncthreads();

        // Stage next tile while PV runs.
        if (jt + 1 < njt) stage(buf ^ 1, jt + 1);

        // PV: every wave handles a 64-wide slice of HEAD_DIM.
#pragma unroll
        for (int jc = 0; jc < 64; jc += 32) {
            BF16x16 a, bm[4];
            a.q[0] = *reinterpret_cast<const i32x4*>(&Ss[l16 * 64 + jc + half * 8]);
            a.q[1] = *reinterpret_cast<const i32x4*>(&Ss[l16 * 64 + jc + 16 + half * 8]);
#pragma unroll
            for (int t = 0; t < 4; ++t) {
                int r = w * 64 + t * 16 + l16;
                bm[t].q[0] = *reinterpret_cast<const i32x4*>(&Vb[r * 64 + jc + half * 8]);
                bm[t].q[1] = *reinterpret_cast<const i32x4*>(&Vb[r * 64 + jc + 16 + half * 8]);
            }
#pragma unroll
            for (int t = 0; t < 4; ++t) acc[t] = wmma_bf16(a, bm[t], acc[t]);
        }

        async_wait0();
        __syncthreads();
        buf ^= 1;
    }

    // Sum of squares over d (raw accumulators; per-row scale folded later).
#pragma unroll
    for (int v = 0; v < 8; ++v) {
        float p = 0.0f;
#pragma unroll
        for (int t = 0; t < 4; ++t) p += acc[t][v] * acc[t][v];
        atomicAdd(&sumsq[v + half * 8], p);
    }
    __syncthreads();

    const float ed = __expf(decay);
#pragma unroll
    for (int v = 0; v < 8; ++v) {
        int m = v + half * 8;
        int i = i0 + m;
        // geometric-series mask normalizer: sum_{t=0..i} exp(decay*t)
        float geo   = (1.0f - __expf(decay * (float)(i + 1))) / (1.0f - ed);
        float r     = rsqrtf(geo);
        float srow  = rowsum[m] * r;                       // sum_j qk_ij
        float denom = fmaxf(1.0f, fabsf(srow));
        float scale = r / denom;
        float rms   = rsqrtf(scale * scale * sumsq[m] * (1.0f / HD) + 1e-5f);
        float tot   = scale * rms;
#pragma unroll
        for (int t = 0; t < 4; ++t) {
            int col = h * HD + w * 64 + t * 16 + l16;
            size_t gi = ((size_t)(b * SEQ + i)) * VDIM + col;
            float gv  = bf2f(gbf[gi]);
            float sil = gv / (1.0f + __expf(-gv));         // silu(g)
            gated[gi] = f2bf(sil * acc[t][v] * tot);
        }
    }
}

// ---------------------------------------------------------------------------
// Host-side launch
// ---------------------------------------------------------------------------
extern "C" void kernel_launch(void* const* d_in, const int* in_sizes, int n_in,
                              void* d_out, int out_size, void* d_ws, size_t ws_size,
                              hipStream_t stream) {
    const float* x  = (const float*)d_in[0];
    const float* wq = (const float*)d_in[1];
    const float* wk = (const float*)d_in[2];
    const float* wv = (const float*)d_in[3];
    const float* wg = (const float*)d_in[4];
    const float* wo = (const float*)d_in[5];
    float* out = (float*)d_out;

    char* ws = (char*)d_ws;
    unsigned short* x_bf  = (unsigned short*)(ws + 0);          //  8 MB
    unsigned short* wqT   = (unsigned short*)(ws + 8388608);    //  2 MB
    unsigned short* wkT   = (unsigned short*)(ws + 10485760);   //  2 MB
    unsigned short* wvT   = (unsigned short*)(ws + 12582912);   //  4 MB
    unsigned short* wgT   = (unsigned short*)(ws + 16777216);   //  4 MB
    unsigned short* woT   = (unsigned short*)(ws + 20971520);   //  4 MB
    float*          qf    = (float*)(ws + 25165824);            // 16 MB
    float*          kf    = (float*)(ws + 41943040);            // 16 MB
    unsigned short* qrb   = (unsigned short*)(ws + 58720256);   //  8 MB
    unsigned short* krb   = (unsigned short*)(ws + 67108864);   //  8 MB
    unsigned short* vrb   = (unsigned short*)(ws + 75497472);   // 16 MB
    unsigned short* gbf   = (unsigned short*)(ws + 92274688);   // 16 MB
    unsigned short* gated = (unsigned short*)(ws + 109051904);  // 16 MB

    // 1) precision conversion (weights transposed for B-operand layout)
    conv_bf16<<<(MROWS * EMBED + 255) / 256, 256, 0, stream>>>(x, x_bf, MROWS * EMBED);
    conv_bf16_T<<<(EMBED * EMBED + 255) / 256, 256, 0, stream>>>(wq, wqT, EMBED, EMBED);
    conv_bf16_T<<<(EMBED * EMBED + 255) / 256, 256, 0, stream>>>(wk, wkT, EMBED, EMBED);
    conv_bf16_T<<<(EMBED * VDIM + 255) / 256, 256, 0, stream>>>(wv, wvT, EMBED, VDIM);
    conv_bf16_T<<<(EMBED * VDIM + 255) / 256, 256, 0, stream>>>(wg, wgT, EMBED, VDIM);
    conv_bf16_T<<<(VDIM * EMBED + 255) / 256, 256, 0, stream>>>(wo, woT, VDIM, EMBED);

    // 2) projections (block tile 128x128)
    dim3 g1(EMBED / 128, MROWS / 128);   // N=1024
    dim3 g2(VDIM / 128, MROWS / 128);    // N=2048
    gemm_bf16<0><<<g1, 256, 0, stream>>>(x_bf, wqT, qf, MROWS, EMBED, EMBED);
    gemm_bf16<0><<<g1, 256, 0, stream>>>(x_bf, wkT, kf, MROWS, EMBED, EMBED);
    gemm_bf16<2><<<g2, 256, 0, stream>>>(x_bf, wvT, vrb, MROWS, VDIM, EMBED);
    gemm_bf16<1><<<g2, 256, 0, stream>>>(x_bf, wgT, gbf, MROWS, VDIM, EMBED);

    // 3) rotary (+ k scale), repack to (b,h,s,KD) bf16
    rotary_kernel<<<(BATCH * SEQ * NH * 64) / 256, 256, 0, stream>>>(qf, kf, qrb, krb);

    // 4) retention attention fused with norm + silu gate
    retnet_attn<<<dim3(SEQ / 16, BATCH * NH), 128, 0, stream>>>(qrb, krb, vrb, gbf, gated);

    // 5) output projection -> fp32 d_out
    gemm_bf16<0><<<dim3(EMBED / 128, MROWS / 128), 256, 0, stream>>>(
        gated, woT, out, MROWS, EMBED, VDIM);
}
